// QuantLinear_42674795053346
// MI455X (gfx1250) — matrix operand order repaired and checked by reading, not compiled, not run on hardware
//
#include <hip/hip_runtime.h>
#include <stdint.h>

typedef __attribute__((ext_vector_type(4)))  float        f32x4;
typedef __attribute__((ext_vector_type(8)))  float        f32x8;
typedef __attribute__((ext_vector_type(4)))  __bf16       bf16x4;
typedef __attribute__((ext_vector_type(8)))  __bf16       bf16x8;
typedef __attribute__((ext_vector_type(16))) __bf16       bf16x16;
typedef __attribute__((ext_vector_type(4)))  unsigned int u32x4;

#define KDIM    4096   // IN
#define OUT_DIM 4096
#define M_TOT   8192   // BATCH*SEQ
#define NOUT_N  32
#define KSTEPS  (KDIM / 32)

// ---------------------------------------------------------------------------
// Kernel 0: convert activations x fp32 -> bf16 once (Xb in workspace).
// ---------------------------------------------------------------------------
__global__ __launch_bounds__(256) void cvt_x_kernel(
    const float* __restrict__ X, __bf16* __restrict__ Xb) {
  const size_t i = ((size_t)blockIdx.x * 256 + threadIdx.x) * 8;
  const f32x4 a = *(const f32x4*)(X + i);
  const f32x4 b = *(const f32x4*)(X + i + 4);
  const bf16x4 lo = __builtin_convertvector(a, bf16x4);
  const bf16x4 hi = __builtin_convertvector(b, bf16x4);
  *(bf16x8*)(Xb + i) =
      __builtin_shufflevector(lo, hi, 0, 1, 2, 3, 4, 5, 6, 7);
}

// ---------------------------------------------------------------------------
// Kernel 1: dequantize 4-bit weights -> bf16, stored TRANSPOSED: Wt[n][k].
// ---------------------------------------------------------------------------
__global__ __launch_bounds__(256) void dequant_transpose_kernel(
    const int* __restrict__ qweight, const float* __restrict__ scales,
    const int* __restrict__ zeros, __bf16* __restrict__ Wt) {
  __shared__ __bf16 Ls[32][520];  // 512 + 8 pad
  const int t   = threadIdx.x;
  const int n0  = blockIdx.x * 32;
  const int k80 = blockIdx.y * 64;

#pragma unroll
  for (int i = 0; i < 8; ++i) {
    const int idx = t + i * 256;
    const int r   = idx >> 5;
    const int c   = idx & 31;
    const int n   = n0 + c;
    const int q   = qweight[(size_t)(k80 + r) * OUT_DIM + n];
    const float s = scales[n];
    const int zw  = zeros[n >> 1];
    const float zf = (float)((n & 1) ? ((zw >> 4) & 15) : (zw & 15));
    bf16x8 v;
#pragma unroll
    for (int j = 0; j < 8; ++j) {
      const float q4 = (float)((q >> (4 * j)) & 15);
      v[j] = (__bf16)((q4 - zf) * s);
    }
    *(bf16x8*)(&Ls[c][r * 8]) = v;
  }
  __syncthreads();

  const int c   = t >> 3;
  const int seg = t & 7;
  __bf16* dst = Wt + (size_t)(n0 + c) * KDIM + (size_t)k80 * 8;
#pragma unroll
  for (int j = 0; j < 8; ++j) {
    const int ko = seg * 64 + j * 8;
    *(bf16x8*)(dst + ko) = *(const bf16x8*)(&Ls[c][ko]);
  }
}

// ---------------------------------------------------------------------------
// Kernel 2: scatter override rows: Wt[n][outids[j]] = oweight[j][n]
// ---------------------------------------------------------------------------
__global__ __launch_bounds__(256) void outids_override_kernel(
    const float* __restrict__ oweight, const int* __restrict__ outids,
    __bf16* __restrict__ Wt) {
  const int idx = blockIdx.x * 256 + threadIdx.x;
  const int j = idx >> 12;
  const int n = idx & (OUT_DIM - 1);
  const int k = outids[j];
  Wt[(size_t)n * KDIM + k] = (__bf16)oweight[(size_t)j * OUT_DIM + n];
}

// ---------------------------------------------------------------------------
// Kernel 3: out = Xb @ Wt^T + bias.
// Block tile 128(M) x 256(N), BK = 32, 256 threads = 8 waves in 2(M) x 4(N).
// Wave tile 64x64 = 4x4 subtiles -> 16 v_wmma per K-step vs 16 ds_load_b128.
// Data movement: 3-deep LDS pipeline filled by GLOBAL_LOAD_ASYNC_TO_LDS_B128
// (ASYNCcnt), 6 async ops per wave per tile, s_wait_asynccnt 6 in steady
// state, one barrier per K-step. No staging registers, no ds_store.
// ---------------------------------------------------------------------------
__global__ __launch_bounds__(256, 1) void gemm_bf16_wmma_kernel(
    const __bf16* __restrict__ Xb, const __bf16* __restrict__ Wt,
    const float* __restrict__ bias, float* __restrict__ out) {
  __shared__ __align__(16) __bf16 As[3][128 * 32];  // 3 x 8 KB
  __shared__ __align__(16) __bf16 Bs[3][256 * 32];  // 3 x 16 KB

  const int tid  = threadIdx.x;
  const int m0   = blockIdx.y * 128;
  const int n0   = blockIdx.x * 256;
  const int wave = tid >> 5;
  const int lane = tid & 31;
  const int lrow = lane & 15;
  const int lhi  = lane >> 4;
  const int wm   = (wave >> 2) * 64;  // 2 waves in M
  const int wn   = (wave & 3) * 64;   // 4 waves in N

  f32x8 zacc;
#pragma unroll
  for (int e = 0; e < 8; ++e) zacc[e] = 0.f;
  f32x8 acc[4][4];
#pragma unroll
  for (int i = 0; i < 4; ++i)
#pragma unroll
    for (int j = 0; j < 4; ++j) acc[i][j] = zacc;

  const __bf16* Ag = Xb + (size_t)m0 * KDIM;
  const __bf16* Bg = Wt + (size_t)n0 * KDIM;

  // Cooperative copy coordinates (b128 chunks; 4 chunks per 32-k row).
  int arow[2], akv[2];
#pragma unroll
  for (int i = 0; i < 2; ++i) { const int c = tid + i * 256; arow[i] = c >> 2; akv[i] = c & 3; }
  int brow[4], bkv[4];
#pragma unroll
  for (int i = 0; i < 4; ++i) { const int c = tid + i * 256; brow[i] = c >> 2; bkv[i] = c & 3; }

  // Issue one K-tile's async memory->LDS copies (6 b128 ops per wave).
  auto issue_tile = [&](int tile, int buf) {
    const int ko = tile * 32;
#pragma unroll
    for (int i = 0; i < 2; ++i) {
      const unsigned long long g = (unsigned long long)(uintptr_t)(
          Ag + (size_t)arow[i] * KDIM + ko + akv[i] * 8);
      const unsigned int l =
          (unsigned int)(uintptr_t)&As[buf][arow[i] * 32 + akv[i] * 8];
      asm volatile("global_load_async_to_lds_b128 %0, %1, off"
                   :: "v"(l), "v"(g) : "memory");
    }
#pragma unroll
    for (int i = 0; i < 4; ++i) {
      const unsigned long long g = (unsigned long long)(uintptr_t)(
          Bg + (size_t)brow[i] * KDIM + ko + bkv[i] * 8);
      const unsigned int l =
          (unsigned int)(uintptr_t)&Bs[buf][brow[i] * 32 + bkv[i] * 8];
      asm volatile("global_load_async_to_lds_b128 %0, %1, off"
                   :: "v"(l), "v"(g) : "memory");
    }
  };

  // Prologue: tiles 0 and 1 in flight.
  issue_tile(0, 0);
  issue_tile(1, 1);

  for (int kt = 0; kt < KSTEPS; ++kt) {
    const int cur = kt % 3;

    // Wait for this wave's tile-kt copies (in-order completion: leaving
    // tile kt+1's 6 ops outstanding means tile kt has landed).
    if (kt + 1 < KSTEPS)
      asm volatile("s_wait_asynccnt 0x6" ::: "memory");
    else
      asm volatile("s_wait_asynccnt 0x0" ::: "memory");
    // Publish tile kt to all waves; also guarantees everyone is done
    // reading buf[(kt-1)%3] == buf[(kt+2)%3], which we refill next.
    __syncthreads();

    if (kt + 2 < KSTEPS) issue_tile(kt + 2, (kt + 2) % 3);

    // A fragments: lane = row lrow, K-chunks at lhi*8 and 16+lhi*8.
    bf16x16 af[4];
#pragma unroll
    for (int mt = 0; mt < 4; ++mt) {
      const __bf16* p = &As[cur][(wm + mt * 16 + lrow) * 32 + lhi * 8];
      const bf16x8 lo = *(const bf16x8*)p;
      const bf16x8 hi = *(const bf16x8*)(p + 16);
      af[mt] = __builtin_shufflevector(lo, hi, 0, 1, 2, 3, 4, 5, 6, 7, 8, 9,
                                       10, 11, 12, 13, 14, 15);
    }

    // B fragments: lane = col lrow, K-chunks at lhi*8 and 16+lhi*8.
#pragma unroll
    for (int nt = 0; nt < 4; ++nt) {
      const __bf16* p = &Bs[cur][(wn + nt * 16 + lrow) * 32 + lhi * 8];
      const bf16x8 lo = *(const bf16x8*)p;
      const bf16x8 hi = *(const bf16x8*)(p + 16);
      const bf16x16 bfr = __builtin_shufflevector(
          lo, hi, 0, 1, 2, 3, 4, 5, 6, 7, 8, 9, 10, 11, 12, 13, 14, 15);
#pragma unroll
      for (int mt = 0; mt < 4; ++mt)
        acc[mt][nt] = __builtin_amdgcn_wmma_f32_16x16x32_bf16(
            false, af[mt], false, bfr, (short)0, acc[mt][nt], false, false);
    }
  }

  // Epilogue: C/D layout -> VGPR r holds (M = r + lhi*8, N = lrow).
#pragma unroll
  for (int nt = 0; nt < 4; ++nt) {
    const int col  = n0 + wn + nt * 16 + lrow;
    const float bv = bias[col];
#pragma unroll
    for (int mt = 0; mt < 4; ++mt) {
      const int rbase = m0 + wm + mt * 16 + lhi * 8;
#pragma unroll
      for (int r = 0; r < 8; ++r)
        out[(size_t)(rbase + r) * OUT_DIM + col] = acc[mt][nt][r] + bv;
    }
  }
}

// ---------------------------------------------------------------------------
extern "C" void kernel_launch(void* const* d_in, const int* in_sizes, int n_in,
                              void* d_out, int out_size, void* d_ws, size_t ws_size,
                              hipStream_t stream) {
  const float* x       = (const float*)d_in[0];
  const int*   qweight = (const int*)d_in[1];
  const float* scales  = (const float*)d_in[2];
  const int*   zeros   = (const int*)d_in[3];
  const float* oweight = (const float*)d_in[4];
  const int*   outids  = (const int*)d_in[5];
  const float* bias    = (const float*)d_in[6];
  float* out = (float*)d_out;

  // Workspace layout: Wt (32 MB bf16, transposed weights) then Xb (64 MB bf16).
  __bf16* Wt = (__bf16*)d_ws;
  __bf16* Xb = Wt + (size_t)OUT_DIM * KDIM;

  cvt_x_kernel<<<(M_TOT * (size_t)KDIM) / 8 / 256, 256, 0, stream>>>(x, Xb);
  dequant_transpose_kernel<<<dim3(OUT_DIM / 32, (KDIM / 8) / 64), 256, 0,
                             stream>>>(qweight, scales, zeros, Wt);
  outids_override_kernel<<<(NOUT_N * OUT_DIM) / 256, 256, 0, stream>>>(
      oweight, outids, Wt);
  gemm_bf16_wmma_kernel<<<dim3(OUT_DIM / 256, M_TOT / 128), 256, 0, stream>>>(
      Xb, Wt, bias, out);
}
